// ScaleDotProductAttention_63617055589269
// MI455X (gfx1250) — matrix operand order repaired and checked by reading, not compiled, not run on hardware
//
#include <hip/hip_runtime.h>
#include <math.h>

// Flash attention fwd for MI455X (gfx1250, wave32), B=4 H=16 S=2048 D=64 fp32.
// f16 WMMA (v_wmma_f32_16x16x32_f16) with f32 accumulation; log2-domain softmax;
// row-max entirely in raw v_max_num_f32 (+DPP ROW_XMASK) asm; row-sum via
// ones-column WMMA; software-pipelined global->reg->LDS staging.

typedef __attribute__((ext_vector_type(16))) _Float16 v16h;
typedef __attribute__((ext_vector_type(8)))  float    v8f;

#define B_DIM 4
#define H_DIM 16
#define S_DIM 2048
#define D_DIM 64

#define WAVES_PER_BLOCK 8
#define THREADS_PER_BLOCK 256
#define M_TILES 2                                          // 16-row q tiles per wave
#define QROWS_PER_WAVE (16 * M_TILES)                      // 32
#define QROWS_PER_BLOCK (WAVES_PER_BLOCK * QROWS_PER_WAVE) // 256
#define QTILES (S_DIM / QROWS_PER_BLOCK)                   // 8
#define KEYS_PER_ITER 32
#define KB_ITERS (S_DIM / KEYS_PER_ITER)                   // 64

#define KV_STRIDE 72   // K tile row stride in halfs (64 + pad)
#define VT_STRIDE 40   // transposed-V row stride in halfs (32 keys + pad)
#define P_STRIDE  40   // P scratch row stride in halfs (32 + pad)

#define LOG2E 1.4426950408889634f

union PackH2 { _Float16 h[2]; unsigned u; };

// mnew = max(mold, rowmax16(max(a,b))) — one asm block of raw v_max ops:
// pair-combine, 4 DPP ROW_XMASK butterfly stages (stay inside each 16-lane
// group), then merge with the running max. No canonicalize instructions.
__device__ __forceinline__ float rowmax_update(float a, float b, float mold) {
  float r;
  asm("v_max_num_f32 %0, %1, %2\n\t"
      "v_max_num_f32_dpp %0, %0, %0 row_xmask:1 row_mask:0xf bank_mask:0xf bound_ctrl:1\n\t"
      "v_max_num_f32_dpp %0, %0, %0 row_xmask:2 row_mask:0xf bank_mask:0xf bound_ctrl:1\n\t"
      "v_max_num_f32_dpp %0, %0, %0 row_xmask:4 row_mask:0xf bank_mask:0xf bound_ctrl:1\n\t"
      "v_max_num_f32_dpp %0, %0, %0 row_xmask:8 row_mask:0xf bank_mask:0xf bound_ctrl:1\n\t"
      "v_max_num_f32 %0, %0, %3"
      : "=&v"(r) : "v"(a), "v"(b), "v"(mold));
  return r;
}

struct KVRegs { float4 ka, kb4; float vv[8]; };

__device__ __forceinline__ void load_tile(const float* __restrict__ kg,
                                          const float* __restrict__ vg,
                                          int keybase, int srow, int sd0,
                                          int vd, int vkp, KVRegs& r) {
  const float* ksrc = kg + (size_t)(keybase + srow) * D_DIM + sd0;
  r.ka  = *reinterpret_cast<const float4*>(ksrc);
  r.kb4 = *reinterpret_cast<const float4*>(ksrc + 4);
  const float* vsrc = vg + (size_t)(keybase + vkp * 8) * D_DIM + vd;
  #pragma unroll
  for (int j = 0; j < 8; ++j) r.vv[j] = vsrc[j * D_DIM];
  // Prefetch one more tile ahead (global_prefetch_b8).
  __builtin_prefetch(ksrc + KEYS_PER_ITER * D_DIM, 0, 1);
  __builtin_prefetch(vsrc + KEYS_PER_ITER * D_DIM, 0, 1);
}

__global__ __launch_bounds__(THREADS_PER_BLOCK)
void sdpa_fwd_f16wmma(const float* __restrict__ q,
                      const float* __restrict__ k,
                      const float* __restrict__ v,
                      float* __restrict__ out) {
  __shared__ _Float16 Klds[KEYS_PER_ITER * KV_STRIDE];
  __shared__ _Float16 Vtlds[D_DIM * VT_STRIDE];                        // V transposed: [d][key]
  __shared__ _Float16 Plds[WAVES_PER_BLOCK * M_TILES * 16 * P_STRIDE];

  const int tid  = threadIdx.x;
  const int wave = tid >> 5;
  const int lane = tid & 31;
  const int half = lane >> 4;
  const int l16  = lane & 15;

  const int bh    = blockIdx.x / QTILES;
  const int qt    = blockIdx.x % QTILES;
  const int qbase = qt * QROWS_PER_BLOCK + wave * QROWS_PER_WAVE;

  const size_t head_off = (size_t)bh * S_DIM * D_DIM;
  const float* qg = q + head_off;
  const float* kg = k + head_off;
  const float* vg = v + head_off;
  float*       og = out + head_off;

  // ---- Q fragments (A layout, 16-bit 16x32): one contiguous 16-elem run per lane (k = 16*half + e) -> 4 float4.
  const float qscale = 0.125f * LOG2E;   // 1/sqrt(D) * log2(e)
  v16h aq[M_TILES][2];
  #pragma unroll
  for (int m2 = 0; m2 < M_TILES; ++m2) {
    const float* qrow = qg + (size_t)(qbase + m2 * 16 + l16) * D_DIM;
    #pragma unroll
    for (int kc = 0; kc < 2; ++kc) {
      const float* base = qrow + kc * 32 + half * 16;
      float4 r0 = *reinterpret_cast<const float4*>(base);
      float4 r1 = *reinterpret_cast<const float4*>(base + 4);
      float4 r2 = *reinterpret_cast<const float4*>(base + 8);
      float4 r3 = *reinterpret_cast<const float4*>(base + 12);
      aq[m2][kc][0]  = (_Float16)(r0.x * qscale);
      aq[m2][kc][1]  = (_Float16)(r0.y * qscale);
      aq[m2][kc][2]  = (_Float16)(r0.z * qscale);
      aq[m2][kc][3]  = (_Float16)(r0.w * qscale);
      aq[m2][kc][4]  = (_Float16)(r1.x * qscale);
      aq[m2][kc][5]  = (_Float16)(r1.y * qscale);
      aq[m2][kc][6]  = (_Float16)(r1.z * qscale);
      aq[m2][kc][7]  = (_Float16)(r1.w * qscale);
      aq[m2][kc][8]  = (_Float16)(r2.x * qscale);
      aq[m2][kc][9]  = (_Float16)(r2.y * qscale);
      aq[m2][kc][10] = (_Float16)(r2.z * qscale);
      aq[m2][kc][11] = (_Float16)(r2.w * qscale);
      aq[m2][kc][12] = (_Float16)(r3.x * qscale);
      aq[m2][kc][13] = (_Float16)(r3.y * qscale);
      aq[m2][kc][14] = (_Float16)(r3.z * qscale);
      aq[m2][kc][15] = (_Float16)(r3.w * qscale);
    }
  }

  v16h ones;   // all-ones B fragment: every output column = row sum of P
  #pragma unroll
  for (int e = 0; e < 16; ++e) ones[e] = (_Float16)1.0f;

  float mrow[M_TILES][8];
  v8f   acc[M_TILES][4];
  v8f   accl[M_TILES];
  #pragma unroll
  for (int m2 = 0; m2 < M_TILES; ++m2) {
    #pragma unroll
    for (int i = 0; i < 8; ++i) mrow[m2][i] = -INFINITY;
    #pragma unroll
    for (int t = 0; t < 4; ++t) acc[m2][t] = (v8f)0.0f;
    accl[m2] = (v8f)0.0f;
  }

  // Staging maps: K row-major (srow,sd0); V transposed (vd, vkp).
  const int srow = tid >> 3;          // 0..31
  const int sd0  = (tid & 7) * 8;     // 0..56
  const int vd   = tid & 63;          // 0..63
  const int vkp  = tid >> 6;          // 0..3

  // Software pipeline: tile kb lives in regs before the loop body stores it.
  KVRegs cur, nxt;
  load_tile(kg, vg, 0, srow, sd0, vd, vkp, cur);

  for (int kb = 0; kb < KB_ITERS; ++kb) {
    // ---- regs -> LDS (K row-major, V transposed), f32->f16 ----
    {
      unsigned* kdst = reinterpret_cast<unsigned*>(&Klds[srow * KV_STRIDE + sd0]);
      PackH2 p;
      p.h[0] = (_Float16)cur.ka.x;  p.h[1] = (_Float16)cur.ka.y;  kdst[0] = p.u;
      p.h[0] = (_Float16)cur.ka.z;  p.h[1] = (_Float16)cur.ka.w;  kdst[1] = p.u;
      p.h[0] = (_Float16)cur.kb4.x; p.h[1] = (_Float16)cur.kb4.y; kdst[2] = p.u;
      p.h[0] = (_Float16)cur.kb4.z; p.h[1] = (_Float16)cur.kb4.w; kdst[3] = p.u;
      unsigned* vdst = reinterpret_cast<unsigned*>(&Vtlds[vd * VT_STRIDE + vkp * 8]);
      #pragma unroll
      for (int j = 0; j < 4; ++j) {
        p.h[0] = (_Float16)cur.vv[2 * j]; p.h[1] = (_Float16)cur.vv[2 * j + 1];
        vdst[j] = p.u;
      }
    }
    __syncthreads();

    // Issue next tile's global loads now; latency overlaps the compute below.
    if (kb + 1 < KB_ITERS)
      load_tile(kg, vg, (kb + 1) * KEYS_PER_ITER, srow, sd0, vd, vkp, nxt);

    // ---- Scores: S[m2][n] = Q[m2] . K^T(n), K fragments shared by both m2 ----
    v8f sfrag[M_TILES][2];
    #pragma unroll
    for (int n = 0; n < 2; ++n) {
      v16h bk[2];
      #pragma unroll
      for (int kc = 0; kc < 2; ++kc) {
        const _Float16* krow = &Klds[(n * 16 + l16) * KV_STRIDE + kc * 32 + half * 16];
        #pragma unroll
        for (int e = 0; e < 16; ++e) bk[kc][e] = krow[e];
      }
      #pragma unroll
      for (int m2 = 0; m2 < M_TILES; ++m2) {
        v8f c = (v8f)0.0f;
        c = __builtin_amdgcn_wmma_f32_16x16x32_f16(false, aq[m2][0], false, bk[0],
                                                   (short)0, c, false, false);
        c = __builtin_amdgcn_wmma_f32_16x16x32_f16(false, aq[m2][1], false, bk[1],
                                                   (short)0, c, false, false);
        sfrag[m2][n] = c;
      }
    }

    // ---- Online softmax (log2 domain; rows M = half*8+i) + P to LDS ----
    #pragma unroll
    for (int m2 = 0; m2 < M_TILES; ++m2) {
      _Float16* pm = &Plds[(wave * M_TILES + m2) * 16 * P_STRIDE];
      #pragma unroll
      for (int i = 0; i < 8; ++i) {
        float mnew = rowmax_update(sfrag[m2][0][i], sfrag[m2][1][i], mrow[m2][i]);
        float p0 = __builtin_amdgcn_exp2f(sfrag[m2][0][i] - mnew);
        float p1 = __builtin_amdgcn_exp2f(sfrag[m2][1][i] - mnew);
        float corr = __builtin_amdgcn_exp2f(mrow[m2][i] - mnew);  // 0 on first pass
        mrow[m2][i] = mnew;
        #pragma unroll
        for (int t = 0; t < 4; ++t) acc[m2][t][i] *= corr;
        accl[m2][i] *= corr;
        int prow = half * 8 + i;
        pm[prow * P_STRIDE +  0 + l16] = (_Float16)p0;
        pm[prow * P_STRIDE + 16 + l16] = (_Float16)p1;
      }
    }

    // ---- Reload P in A-fragment layout (same-wave LDS RAW, dscnt-ordered) ----
    v16h ap[M_TILES];
    #pragma unroll
    for (int m2 = 0; m2 < M_TILES; ++m2) {
      const _Float16* prow = &Plds[(wave * M_TILES + m2) * 16 * P_STRIDE + l16 * P_STRIDE];
      #pragma unroll
      for (int e = 0; e < 16; ++e) {
        int kk = half * 16 + e;
        ap[m2][e] = prow[kk];
      }
    }

    // ---- O += P . V : 4 d-tiles, V fragment shared by both m2 ----
    #pragma unroll
    for (int t = 0; t < 4; ++t) {
      v16h bv;   // b[e]: K = key = half*16+e, N = d = t*16+l16 (from Vt[d][key])
      const _Float16* vrow = &Vtlds[(t * 16 + l16) * VT_STRIDE + half * 16];
      #pragma unroll
      for (int e = 0; e < 16; ++e) bv[e] = vrow[e];
      #pragma unroll
      for (int m2 = 0; m2 < M_TILES; ++m2)
        acc[m2][t] = __builtin_amdgcn_wmma_f32_16x16x32_f16(false, ap[m2], false, bv,
                                                            (short)0, acc[m2][t], false, false);
    }
    // ---- l += P . 1 (row sums broadcast to every lane of the half-group) ----
    #pragma unroll
    for (int m2 = 0; m2 < M_TILES; ++m2)
      accl[m2] = __builtin_amdgcn_wmma_f32_16x16x32_f16(false, ap[m2], false, ones,
                                                        (short)0, accl[m2], false, false);

    cur = nxt;
    __syncthreads();   // all waves done with K/V before restaging
  }

  // ---- Epilogue: O = acc / l ----
  #pragma unroll
  for (int m2 = 0; m2 < M_TILES; ++m2) {
    #pragma unroll
    for (int i = 0; i < 8; ++i) {
      float inv_l = 1.0f / accl[m2][i];
      float* orow = og + (size_t)(qbase + m2 * 16 + half * 8 + i) * D_DIM;
      #pragma unroll
      for (int t = 0; t < 4; ++t)
        orow[t * 16 + l16] = acc[m2][t][i] * inv_l;
    }
  }
}

extern "C" void kernel_launch(void* const* d_in, const int* in_sizes, int n_in,
                              void* d_out, int out_size, void* d_ws, size_t ws_size,
                              hipStream_t stream) {
  (void)in_sizes; (void)n_in; (void)out_size; (void)d_ws; (void)ws_size;
  const float* q = (const float*)d_in[0];
  const float* k = (const float*)d_in[1];
  const float* v = (const float*)d_in[2];
  float* out = (float*)d_out;

  dim3 grid(B_DIM * H_DIM * QTILES);    // 512 blocks
  dim3 block(THREADS_PER_BLOCK);        // 256 threads = 8 waves
  sdpa_fwd_f16wmma<<<grid, block, 0, stream>>>(q, k, v, out);
}
